// GATFeatureOptimizer_51376398795418
// MI455X (gfx1250) — matrix operand order repaired and checked by reading, not compiled, not run on hardware
//
#include <hip/hip_runtime.h>

#define ALPHA 0.2f
#define LOG2E 1.4426950408889634f
#define BB    4096
#define NN    10000
#define DD    256
#define NPAD  10016            // next multiple of 32 above N
#define NCHUNK (NPAD / 32)     // 313

typedef __attribute__((ext_vector_type(16))) _Float16 v16h;
typedef __attribute__((ext_vector_type(8)))  _Float16 v8h;
typedef __attribute__((ext_vector_type(8)))  float    v8f;

// ---------------------------------------------------------------------------
// score[r] = dot(X[r,:], W) * log2(e)  (one wave32 per row).
// Rows >= validRows -> -1e30 (softmax weight exactly 0 for padded columns).
// Scaling by log2(e) is exact for the softmax since c>0 => c*leaky(x)=leaky(c*x).
// ---------------------------------------------------------------------------
__global__ __launch_bounds__(256) void score_rows_kernel(
    const float* __restrict__ X, const float* __restrict__ W,
    float* __restrict__ out, int validRows, int totalRows) {
  int wave = (int)((blockIdx.x * blockDim.x + threadIdx.x) >> 5);
  int lane = threadIdx.x & 31;
  if (wave >= totalRows) return;
  float s;
  if (wave < validRows) {
    const float* x = X + (size_t)wave * DD;
    s = 0.f;
#pragma unroll
    for (int i = 0; i < DD / 32; ++i) s += x[lane + 32 * i] * W[lane + 32 * i];
#pragma unroll
    for (int off = 16; off > 0; off >>= 1) s += __shfl_xor(s, off, 32);
    s *= LOG2E;
  } else {
    s = -1e30f;
  }
  if (lane == 0) out[wave] = s;
}

// ---------------------------------------------------------------------------
// Vt[d][n] = (f16) disease_nodes[n][d], zero-padded to NPAD columns.
// ---------------------------------------------------------------------------
__global__ __launch_bounds__(256) void transpose_cvt_kernel(
    const float* __restrict__ dn, _Float16* __restrict__ Vt) {
  __shared__ float tile[32][33];
  int d0 = blockIdx.x * 32;
  int n0 = blockIdx.y * 32;
  int tx = threadIdx.x;  // 0..31
  int ty = threadIdx.y;  // 0..7
#pragma unroll
  for (int i = 0; i < 32; i += 8) {
    int n = n0 + ty + i;
    tile[ty + i][tx] = (n < NN) ? dn[(size_t)n * DD + d0 + tx] : 0.f;
  }
  __syncthreads();
#pragma unroll
  for (int i = 0; i < 32; i += 8) {
    int d = d0 + ty + i;
    Vt[(size_t)d * NPAD + n0 + tx] = (_Float16)tile[tx][ty + i];
  }
}

// ---------------------------------------------------------------------------
// Global max of (scaled) score_n over the valid range (single block).
// ---------------------------------------------------------------------------
__global__ __launch_bounds__(256) void max_reduce_kernel(
    const float* __restrict__ sn, float* __restrict__ out) {
  __shared__ float red[256];
  float m = -1e30f;
  for (int i = threadIdx.x; i < NN; i += 256) m = fmaxf(m, sn[i]);
  red[threadIdx.x] = m;
  __syncthreads();
  for (int s = 128; s > 0; s >>= 1) {
    if ((int)threadIdx.x < s)
      red[threadIdx.x] = fmaxf(red[threadIdx.x], red[threadIdx.x + s]);
    __syncthreads();
  }
  if (threadIdx.x == 0) *out = red[0];
}

// ---------------------------------------------------------------------------
// Main: one block per 16-row patient tile; 8 waves split the N chunks.
// Exact softmax max (log2 domain): m_r = leaky(sp2[r] + max_n sn2).
// agg accumulated with v_wmma_f32_16x16x32_f16; cross-wave combine in LDS.
// ---------------------------------------------------------------------------
__global__ __launch_bounds__(256) void gat_wmma_kernel(
    const float* __restrict__ pf, const _Float16* __restrict__ Vt,
    const float* __restrict__ score_n, const float* __restrict__ score_p,
    const float* __restrict__ max_sn, float* __restrict__ out) {
  extern __shared__ float smem[];
  float* accS = smem;             // 8 waves * 16x256 fp32 = 128 KB
  float* rsS  = smem + 8 * 4096;  // 8 waves * 16 row sums

  const int tid  = threadIdx.x;
  const int lane = tid & 31;
  const int w    = __builtin_amdgcn_readfirstlane(tid >> 5);  // scalar wave id
  const int row0 = blockIdx.x * 16;
  const int r    = lane & 15;          // patient row within tile
  const int hi   = (lane >> 4) & 1;    // half-wave index

  const float sp   = score_p[row0 + r];          // already * log2(e)
  const float mraw = sp + *max_sn;
  const float m    = fmaxf(mraw, ALPHA * mraw);  // exact row max, log2 domain

  v8f acc[16];
  const v8f vzero = {0.f, 0.f, 0.f, 0.f, 0.f, 0.f, 0.f, 0.f};
#pragma unroll
  for (int t = 0; t < 16; ++t) acc[t] = vzero;
  float rsum0 = 0.f, rsum1 = 0.f, rsum2 = 0.f, rsum3 = 0.f;

  const int aoff = hi * 8;    // A-fragment K offset (ISA 16-bit A 16x32 layout)
  const int boff = hi * 8;    // B-fragment K offset (same K layout as A)
  const _Float16* vbase = Vt + (size_t)(lane & 15) * NPAD;

  for (int c = w; c < NCHUNK; c += 8) {
    const int n0 = c * 32;
    const float sn = score_n[n0 + lane];
    const _Float16* bp = vbase + n0 + boff;

    // Kick off B loads for tiles 0 and 1 before the long A-build chain.
    v8h b0a = *(const v8h*)(bp);
    v8h b0b = *(const v8h*)(bp + 16);
    v8h b1a = *(const v8h*)(bp + (size_t)16 * NPAD);
    v8h b1b = *(const v8h*)(bp + (size_t)16 * NPAD + 16);

    // Build A fragment: probs for (row r, K-slot pattern of this lane)
    v16h a;
#pragma unroll
    for (int h = 0; h < 16; ++h) {
      int k  = h + ((h >= 8) ? 8 : 0) + aoff;     // K index held in half h
      float s = sp + __shfl(sn, k, 32);           // ds_bpermute
      s = fmaxf(s, ALPHA * s);                    // leaky relu (both signs)
      float e = __builtin_amdgcn_exp2f(s - m);    // bare v_exp_f32
      if ((h & 3) == 0)      rsum0 += e;
      else if ((h & 3) == 1) rsum1 += e;
      else if ((h & 3) == 2) rsum2 += e;
      else                   rsum3 += e;
      a[h] = (_Float16)e;
    }

    // 16 D-tiles, software-pipelined two tiles deep.
#pragma unroll
    for (int t = 0; t < 16; ++t) {
      v8h na = b1a, nb = b1b;
      if (t < 14) {
        const _Float16* np = bp + (size_t)(t + 2) * 16 * NPAD;
        na = *(const v8h*)np;
        nb = *(const v8h*)(np + 16);
      }
      v16h bf = __builtin_shufflevector(b0a, b0b, 0, 1, 2, 3, 4, 5, 6, 7,
                                        8, 9, 10, 11, 12, 13, 14, 15);
      acc[t] = __builtin_amdgcn_wmma_f32_16x16x32_f16(
          false, a, false, bf, (short)0, acc[t], false, false);
      b0a = b1a; b0b = b1b;
      b1a = na;  b1b = nb;
    }
  }

  // Row sum: lanes r and r+16 partition K, so one xor-16 reduce
  float rowsum = (rsum0 + rsum1) + (rsum2 + rsum3);
  const float rs = rowsum + __shfl_xor(rowsum, 16, 32);

  // Spill C fragments to LDS (C layout: VGPR i -> row i / i+8, col = lane&15)
  const int crow = hi * 8;
#pragma unroll
  for (int t = 0; t < 16; ++t) {
#pragma unroll
    for (int i = 0; i < 8; ++i)
      accS[w * 4096 + (crow + i) * 256 + t * 16 + (lane & 15)] = acc[t][i];
  }
  if (lane < 16) rsS[w * 16 + lane] = rs;
  __syncthreads();

  // Combine 8 waves, normalize, add residual. Thread tid owns column d=tid.
#pragma unroll
  for (int i = 0; i < 16; ++i) {
    float a = 0.f, rsum = 0.f;
#pragma unroll
    for (int w8 = 0; w8 < 8; ++w8) {
      a    += accS[w8 * 4096 + i * 256 + tid];
      rsum += rsS[w8 * 16 + i];
    }
    const size_t o = (size_t)(row0 + i) * DD + tid;
    out[o] = pf[o] + a / rsum;
  }
}

// ---------------------------------------------------------------------------
extern "C" void kernel_launch(void* const* d_in, const int* in_sizes, int n_in,
                              void* d_out, int out_size, void* d_ws, size_t ws_size,
                              hipStream_t stream) {
  const float* pf = (const float*)d_in[0];  // patient_features (4096 x 256)
  const float* dn = (const float*)d_in[1];  // disease_nodes   (10000 x 256)
  const float* ak = (const float*)d_in[2];  // attn_kernel     (512 x 1)
  float* out = (float*)d_out;

  char* ws = (char*)d_ws;
  _Float16* Vt  = (_Float16*)ws;                         // 256 * NPAD f16
  float*    sn  = (float*)(ws + (size_t)DD * NPAD * 2);  // NPAD
  float*    sp  = sn + NPAD;                             // 4096
  float*    msn = sp + BB;                               // 1

  score_rows_kernel<<<BB / 8, 256, 0, stream>>>(pf, ak, sp, BB, BB);
  score_rows_kernel<<<NPAD / 8, 256, 0, stream>>>(dn, ak + DD, sn, NN, NPAD);
  transpose_cvt_kernel<<<dim3(DD / 32, NPAD / 32), dim3(32, 8), 0, stream>>>(dn, Vt);
  max_reduce_kernel<<<1, 256, 0, stream>>>(sn, msn);

  const size_t ldsBytes = (8 * 4096 + 8 * 16) * sizeof(float);  // 131584 B
  gat_wmma_kernel<<<BB / 16, 256, ldsBytes, stream>>>(pf, Vt, sn, sp, msn, out);
}
